// _MXFPLinearBase_27032524161118
// MI455X (gfx1250) — matrix operand order repaired and checked by reading, not compiled, not run on hardware
//
#include <hip/hip_runtime.h>

// CDNA5 / gfx1250 vector types for WMMA
typedef __attribute__((ext_vector_type(16))) _Float16 v16h;
typedef __attribute__((ext_vector_type(8)))  _Float16 v8h;
typedef __attribute__((ext_vector_type(8)))  float    v8f;

// Pointer element type expected by the async-to-LDS builtins on this
// toolchain (per hipcc diagnostic: 'int __attribute__((vector_size(16))) *').
typedef int i128v __attribute__((vector_size(16)));

#define MXBLK 32

// gfx1250 async global->LDS copies (ASYNCcnt-tracked), guarded so the file
// still compiles on toolchains that lack the builtins.
#if defined(__has_builtin)
#if __has_builtin(__builtin_amdgcn_global_load_async_to_lds_b128)
#define HAVE_ASYNC_LDS 1
#endif
#if __has_builtin(__builtin_amdgcn_s_wait_asynccnt)
#define HAVE_WAIT_ASYNC 1
#endif
#endif

// Fence for ASYNCcnt: builtin when available, raw gfx1250 asm otherwise.
__device__ __forceinline__ void wait_async_zero() {
#if defined(HAVE_WAIT_ASYNC)
  __builtin_amdgcn_s_wait_asynccnt(0);
#elif defined(HAVE_ASYNC_LDS)
  asm volatile("s_wait_asynccnt 0x0" ::: "memory");
#endif
}

// Round |x|/s (already normalized) to nearest E2M1 magnitude, using the
// reference's searchsorted(BOUNDS) midpoint boundaries (ties go down).
__device__ __forceinline__ float quant_e2m1_mag(float a) {
  a = fminf(a, 6.0f);
  float q;
  if      (a <= 0.25f) q = 0.0f;
  else if (a <= 0.75f) q = 0.5f;
  else if (a <= 1.25f) q = 1.0f;
  else if (a <= 1.75f) q = 1.5f;
  else if (a <= 2.5f ) q = 2.0f;
  else if (a <= 3.5f ) q = 3.0f;
  else if (a <= 5.0f ) q = 4.0f;
  else                 q = 6.0f;
  return q;
}

// One thread per MX block of 32 contiguous fp32 values.
// Writes 32 f16 quantized values (exact E2M1 levels) + one fp32 scale.
__global__ void mxfp4_quantize(const float* __restrict__ src,
                               _Float16* __restrict__ qout,
                               float* __restrict__ sout,
                               int totalBlocks) {
  int idx = blockIdx.x * blockDim.x + threadIdx.x;
  if (idx >= totalBlocks) return;
  const float4* p = reinterpret_cast<const float4*>(src + (size_t)idx * MXBLK);
  float4 v[8];
  float m = 0.0f;
#pragma unroll
  for (int i = 0; i < 8; ++i) {
    v[i] = p[i];
    m = fmaxf(m, fmaxf(fmaxf(fabsf(v[i].x), fabsf(v[i].y)),
                       fmaxf(fabsf(v[i].z), fabsf(v[i].w))));
  }
  float s = fmaxf(m * (1.0f / 6.0f), 1e-30f);

  __align__(16) _Float16 qh[32];
#pragma unroll
  for (int i = 0; i < 8; ++i) {
    float e0 = v[i].x, e1 = v[i].y, e2 = v[i].z, e3 = v[i].w;
    float t0 = e0 / s, t1 = e1 / s, t2 = e2 / s, t3 = e3 / s;
    qh[i * 4 + 0] = (_Float16)copysignf(quant_e2m1_mag(fabsf(t0)), t0);
    qh[i * 4 + 1] = (_Float16)copysignf(quant_e2m1_mag(fabsf(t1)), t1);
    qh[i * 4 + 2] = (_Float16)copysignf(quant_e2m1_mag(fabsf(t2)), t2);
    qh[i * 4 + 3] = (_Float16)copysignf(quant_e2m1_mag(fabsf(t3)), t3);
  }
  uint4* dst = reinterpret_cast<uint4*>(qout + (size_t)idx * MXBLK);
  const uint4* qsrc = reinterpret_cast<const uint4*>(qh);
#pragma unroll
  for (int i = 0; i < 4; ++i) dst[i] = qsrc[i];
  sout[idx] = s;
}

// 128x128 C tile per workgroup (256 threads = 8 wave32), double-buffered LDS.
// Wave grid 4(M) x 2(N): each wave owns a 32x64 strip = 2x4 subtiles of 16x16.
// K loop walks one MX block (K=32) at a time: one v_wmma_f32_16x16x32_f16 per
// subtile with C=0, then fp32 scale-FMA with s_x[m]*s_w[n] into accumulators.
// Next k-block's tiles are staged with GLOBAL_LOAD_ASYNC_TO_LDS_B128 before
// the WMMAs; s_wait_asynccnt 0 + barrier after them (overlap).
__global__ __launch_bounds__(256)
void mxfp4_gemm(const _Float16* __restrict__ xq, const float* __restrict__ xs,
                const _Float16* __restrict__ wq, const float* __restrict__ wsc,
                const float* __restrict__ bias, float* __restrict__ out,
                int M, int J, int K, int NB) {
  __shared__ __align__(16) _Float16 shA[2][128 * 32];   // [buf][row][k]  2x8 KB
  __shared__ __align__(16) _Float16 shB[2][128 * 32];   // [buf][col][k]  2x8 KB
  __shared__ float shXS[2][128];
  __shared__ float shWS[2][128];

  const int t    = threadIdx.x;
  const int wave = t >> 5;
  const int lane = t & 31;
  const int hf   = lane >> 4;     // lane half (0: lanes 0-15, 1: lanes 16-31)
  const int lr   = lane & 15;
  const int wm   = wave >> 1;     // 0..3 -> 32-row strip
  const int wn   = wave & 1;      // 0..1 -> 64-col strip
  const int mBase = blockIdx.y * 128;
  const int nBase = blockIdx.x * 128;

  // Per-thread staging geometry: 512 16B chunks per tile, 4 chunks per row.
  const int c0 = t;         // chunk for i=0
  const int c1 = t + 256;   // chunk for i=1

  auto stage = [&](int kb, int buf) {
#pragma unroll
    for (int i = 0; i < 2; ++i) {
      int c  = (i == 0) ? c0 : c1;
      int r  = c >> 2;
      int co = (c & 3) << 3;                      // halfword offset within row
      const _Float16* ga = &xq[(size_t)(mBase + r) * K + kb * 32 + co];
      const _Float16* gb = &wq[(size_t)(nBase + r) * K + kb * 32 + co];
      _Float16* la = &shA[buf][r * 32 + co];
      _Float16* lb = &shB[buf][r * 32 + co];
#if defined(HAVE_ASYNC_LDS)
      __builtin_amdgcn_global_load_async_to_lds_b128((i128v*)ga, (i128v*)la, 0, 0);
      __builtin_amdgcn_global_load_async_to_lds_b128((i128v*)gb, (i128v*)lb, 0, 0);
#else
      *reinterpret_cast<uint4*>(la) = *reinterpret_cast<const uint4*>(ga);
      *reinterpret_cast<uint4*>(lb) = *reinterpret_cast<const uint4*>(gb);
#endif
    }
    if (t < 128) shXS[buf][t]       = xs [(size_t)(mBase + t)       * NB + kb];
    else         shWS[buf][t - 128] = wsc[(size_t)(nBase + t - 128) * NB + kb];
  };

  v8f acc[2][4];
  {
    v8f z = {0.f, 0.f, 0.f, 0.f, 0.f, 0.f, 0.f, 0.f};
#pragma unroll
    for (int mi = 0; mi < 2; ++mi)
#pragma unroll
      for (int ni = 0; ni < 4; ++ni) acc[mi][ni] = z;
  }

  // Prologue: stage k-block 0 into buffer 0.
  stage(0, 0);
  wait_async_zero();
  __syncthreads();

  for (int kb = 0; kb < NB; ++kb) {
    const int buf = kb & 1;

    // Kick off async staging of the next k-block into the other buffer.
    if (kb + 1 < NB) stage(kb + 1, buf ^ 1);

    // L2 warm-up two k-blocks ahead (emits global_prefetch_b8).
    if (kb + 2 < NB) {
      if (t < 128) __builtin_prefetch(&xq[(size_t)(mBase + t)       * K + (kb + 2) * 32], 0, 1);
      else         __builtin_prefetch(&wq[(size_t)(nBase + t - 128) * K + (kb + 2) * 32], 0, 1);
    }

    const _Float16* A = shA[buf];
    const _Float16* B = shB[buf];
    const float*    SX = shXS[buf];
    const float*    SW = shWS[buf];

    // ---- build fragments, contiguous k = 16*hf + s in both A and B ----
    v16h afrag[2];
    v16h bfrag[4];
    float sxv[2][8];
    float swv[4];
#pragma unroll
    for (int mi = 0; mi < 2; ++mi) {
      int row = wm * 32 + mi * 16 + lr;
      // A 16x32: lanes 0-15 hold K 0-15, lanes 16-31 hold K 16-31
      afrag[mi] = *reinterpret_cast<const v16h*>(&A[row * 32 + hf * 16]);
#pragma unroll
      for (int r = 0; r < 8; ++r)
        sxv[mi][r] = SX[wm * 32 + mi * 16 + hf * 8 + r];
    }
#pragma unroll
    for (int ni = 0; ni < 4; ++ni) {
      int col = wn * 64 + ni * 16 + lr;
      // B 32x16: column per lane; lanes 0-15 K 0-15, lanes 16-31 K 16-31
      bfrag[ni] = *reinterpret_cast<const v16h*>(&B[col * 32 + hf * 16]);
      swv[ni]   = SW[wn * 64 + ni * 16 + lr];
    }

    // ---- 8 WMMAs + scale-accumulate ----
#pragma unroll
    for (int mi = 0; mi < 2; ++mi) {
#pragma unroll
      for (int ni = 0; ni < 4; ++ni) {
        v8f zero = {0.f, 0.f, 0.f, 0.f, 0.f, 0.f, 0.f, 0.f};
        v8f p = __builtin_amdgcn_wmma_f32_16x16x32_f16(
            false, afrag[mi], false, bfrag[ni], (short)0, zero, false, false);
#pragma unroll
        for (int r = 0; r < 8; ++r)
          acc[mi][ni][r] = fmaf(p[r] * sxv[mi][r], swv[ni], acc[mi][ni][r]);
      }
    }

    // Wait for the async stage of the next buffer, then publish to all waves.
    wait_async_zero();
    __syncthreads();
  }

  // ---- epilogue: C layout is M = r + 8*hf (VGPR r), N = lr ----
#pragma unroll
  for (int ni = 0; ni < 4; ++ni) {
    int col = nBase + wn * 64 + ni * 16 + lr;
    float bv = bias[col];
#pragma unroll
    for (int mi = 0; mi < 2; ++mi) {
      int row0 = mBase + wm * 32 + mi * 16 + hf * 8;
#pragma unroll
      for (int r = 0; r < 8; ++r)
        out[(size_t)(row0 + r) * J + col] = acc[mi][ni][r] + bv;
    }
  }
}

extern "C" void kernel_launch(void* const* d_in, const int* in_sizes, int n_in,
                              void* d_out, int out_size, void* d_ws, size_t ws_size,
                              hipStream_t stream) {
  const float* x    = (const float*)d_in[0];
  const float* w    = (const float*)d_in[1];
  const float* bias = (const float*)d_in[2];
  float* out = (float*)d_out;

  const int J  = in_sizes[2];           // 4096 output features
  const int K  = in_sizes[1] / J;       // 1024 reduction dim
  const int M  = in_sizes[0] / K;       // 2048 rows
  const int NB = K / MXBLK;             // 32 MX blocks per row

  // Workspace layout: xq (f16) | wq (f16) | xs (f32) | ws (f32)
  char* wsb = (char*)d_ws;
  _Float16* xq = (_Float16*)wsb;
  _Float16* wq = (_Float16*)(wsb + (size_t)M * K * sizeof(_Float16));
  float*    xs = (float*)   (wsb + (size_t)(M + J) * K * sizeof(_Float16));
  float*    wsc = xs + (size_t)M * NB;

  const int xb = M * NB;
  const int wb = J * NB;
  mxfp4_quantize<<<(xb + 255) / 256, 256, 0, stream>>>(x, xq, xs, xb);
  mxfp4_quantize<<<(wb + 255) / 256, 256, 0, stream>>>(w, wq, wsc, wb);

  dim3 grid(J / 128, M / 128);
  mxfp4_gemm<<<grid, 256, 0, stream>>>(xq, xs, wq, wsc, bias, out, M, J, K, NB);
}
